// SVDHead_36876589203990
// MI455X (gfx1250) — matrix operand order repaired and checked by reading, not compiled, not run on hardware
//
#include <hip/hip_runtime.h>

typedef __bf16 bf16_t;
typedef __attribute__((ext_vector_type(16))) __bf16 v16bf;
typedef __attribute__((ext_vector_type(8)))  __bf16 v8bf;
typedef __attribute__((ext_vector_type(8)))  float  v8f;

#define BATCH   16
#define DKDIM   512
#define NDIM    2048
#define MDIM    2048
#define ROWBLK  16
#define NBLOCKS (NDIM / ROWBLK)                 /* 128 row-blocks per batch */
#define SCALE   0.044194173824159216f          /* 1/sqrt(512) */

/* ---------------- transpose + f32 -> bf16 convert ----------------
 * in:  [B][DKDIM][L] f32   (L = 2048)
 * out: [B][L][DKDIM] bf16 bits (round-to-nearest-even)
 * so that per-row K is contiguous for WMMA fragment loads.         */
__global__ __launch_bounds__(256) void transpose_cvt_kernel(
    const float* __restrict__ in, unsigned short* __restrict__ out)
{
  __shared__ float tile[32][33];
  const int b  = blockIdx.z;
  const int d0 = blockIdx.y * 32;
  const int n0 = blockIdx.x * 32;
  const int tx = threadIdx.x;
  const int ty = threadIdx.y;

  const float* ip = in + ((size_t)b * DKDIM + d0) * NDIM + n0;
#pragma unroll
  for (int j = 0; j < 32; j += 8)
    tile[ty + j][tx] = ip[(size_t)(ty + j) * NDIM + tx];
  __syncthreads();

  unsigned short* op = out + ((size_t)b * NDIM + n0) * DKDIM + d0;
#pragma unroll
  for (int j = 0; j < 32; j += 8) {
    float v = tile[tx][ty + j];
    unsigned u = __float_as_uint(v);
    unsigned r = (u + 0x7FFFu + ((u >> 16) & 1u)) >> 16;   /* RNE to bf16 */
    op[(size_t)(ty + j) * DKDIM + tx] = (unsigned short)r;
  }
}

/* ---- 16-lane-group reductions (matches WMMA C/D layout: each row of a
 * 16x16 f32 tile lives in one VGPR across a 16-lane half-group). ---- */
__device__ __forceinline__ float red16_max(float v) {
#pragma unroll
  for (int m = 1; m < 16; m <<= 1) v = fmaxf(v, __shfl_xor(v, m, 32));
  return v;
}
__device__ __forceinline__ float red16_sum(float v) {
#pragma unroll
  for (int m = 1; m < 16; m <<= 1) v += __shfl_xor(v, m, 32);
  return v;
}

/* ---------------- fused scores-GEMM + softmax + corr ----------------
 * One workgroup = 16 src rows of one batch, all 2048 target columns.
 * 16 waves; wave w owns columns [128w, 128w+128) = 8 WMMA tiles.
 * k-outer / tile-inner: each A fragment is loaded once per k-step and
 * reused by 8 independent-accumulator WMMAs (2.1 vmem per wmma).
 * Scores never leave registers; softmax stats cross waves via LDS.  */
__global__ __launch_bounds__(512, 1) void fused_softmax_corr_kernel(
    const bf16_t* __restrict__ Qt,   /* [B][N][DK] bf16 */
    const bf16_t* __restrict__ Kt,   /* [B][M][DK] bf16 */
    const float*  __restrict__ tgt,  /* [B][3][M]  f32  */
    const float*  __restrict__ srck, /* [B][3][N]  f32  */
    float*        __restrict__ part) /* [B*NBLOCKS][16] partial sums */
{
  const int b    = blockIdx.y;
  const int n0   = blockIdx.x * ROWBLK;
  const int tid  = threadIdx.x;
  const int lane = tid & 31;
  const int wave = tid >> 5;      /* 0..15 */
  const int l16  = lane & 15;
  const int half = lane >> 4;     /* 0/1   */

  __shared__ float redbuf[16][16];     /* per-wave row partials       */
  __shared__ float corrP[16][16][3];   /* per-wave corr partials      */
  __shared__ float rowmaxg[16];
  __shared__ float rowsumg[16];
  __shared__ float corrW[16][3];

  /* A fragment (16x32 bf16): lane -> row = l16; one contiguous
   * 16-half run at K = k + 16*half (matches B order).                */
  const bf16_t* qrow =
      Qt + ((size_t)b * NDIM + (n0 + l16)) * DKDIM + half * 16;
  const int mwb = wave * 128;

  /* B fragment (32x16 bf16): lane -> column = l16, one contiguous
   * 16-half run at K = k + 16*half (same k order as A).            */
  const bf16_t* kbase =
      Kt + ((size_t)b * MDIM + (mwb + l16)) * DKDIM + half * 16;

  __builtin_prefetch(kbase, 0, 1);

  /* ---- GEMM phase: 8 independent accumulators, A reused across tiles ---- */
  v8f accT[8];
#pragma unroll
  for (int t = 0; t < 8; ++t) accT[t] = (v8f){};

#pragma unroll 2
  for (int k = 0; k < DKDIM; k += 32) {
    v8bf alo = *(const v8bf*)(qrow + k);
    v8bf ahi = *(const v8bf*)(qrow + k + 8);
    v16bf a  = __builtin_shufflevector(alo, ahi,
                 0,1,2,3,4,5,6,7,8,9,10,11,12,13,14,15);
#pragma unroll
    for (int t = 0; t < 8; ++t) {
      v16bf bm = *(const v16bf*)(kbase + (size_t)t * 16 * DKDIM + k);
      accT[t] = __builtin_amdgcn_wmma_f32_16x16x32_bf16(
                    false, a, false, bm, (short)0, accT[t], false, false);
    }
  }

  /* ---- row max: tile regs -> 16-lane shuffle -> LDS across waves ---- */
  {
    float mrow[8];
#pragma unroll
    for (int r = 0; r < 8; ++r) {
      float v = accT[0][r];
#pragma unroll
      for (int t = 1; t < 8; ++t) v = fmaxf(v, accT[t][r]);
      mrow[r] = red16_max(v);
    }
    if (l16 == 0) {
#pragma unroll
      for (int r = 0; r < 8; ++r) redbuf[wave][r + 8 * half] = mrow[r];
    }
  }
  __syncthreads();
  if (tid < 16) {
    float v = redbuf[0][tid];
    for (int w = 1; w < 16; ++w) v = fmaxf(v, redbuf[w][tid]);
    rowmaxg[tid] = v;
  }
  __syncthreads();

  /* ---- exp + row sum + corr accumulation (V = tgt, width 3) ---- */
  {
    float rmax[8];
#pragma unroll
    for (int r = 0; r < 8; ++r) rmax[r] = rowmaxg[r + 8 * half];

    float srow[8], ca0[8], ca1[8], ca2[8];
#pragma unroll
    for (int r = 0; r < 8; ++r) { srow[r] = 0.f; ca0[r] = ca1[r] = ca2[r] = 0.f; }

#pragma unroll
    for (int t = 0; t < 8; ++t) {
      const int m = mwb + t * 16 + l16;
      const float t0 = tgt[((size_t)b * 3 + 0) * MDIM + m];
      const float t1 = tgt[((size_t)b * 3 + 1) * MDIM + m];
      const float t2 = tgt[((size_t)b * 3 + 2) * MDIM + m];
#pragma unroll
      for (int r = 0; r < 8; ++r) {
        float p = __expf((accT[t][r] - rmax[r]) * SCALE);
        srow[r] += p;
        ca0[r] += p * t0; ca1[r] += p * t1; ca2[r] += p * t2;
      }
    }
#pragma unroll
    for (int r = 0; r < 8; ++r) {
      srow[r] = red16_sum(srow[r]);
      ca0[r]  = red16_sum(ca0[r]);
      ca1[r]  = red16_sum(ca1[r]);
      ca2[r]  = red16_sum(ca2[r]);
    }
    if (l16 == 0) {
#pragma unroll
      for (int r = 0; r < 8; ++r) {
        const int row = r + 8 * half;
        redbuf[wave][row]    = srow[r];
        corrP[wave][row][0]  = ca0[r];
        corrP[wave][row][1]  = ca1[r];
        corrP[wave][row][2]  = ca2[r];
      }
    }
  }
  __syncthreads();
  if (tid < 16) {
    float v = 0.f;
    for (int w = 0; w < 16; ++w) v += redbuf[w][tid];
    rowsumg[tid] = v;
  }
  if (tid >= 64 && tid < 112) {
    const int q = tid - 64, row = q / 3, c = q % 3;
    float v = 0.f;
    for (int w = 0; w < 16; ++w) v += corrP[w][row][c];
    corrW[row][c] = v;
  }
  __syncthreads();

  /* ---- per-block Kabsch partials: 9 of src*corr^T, 3 src sums, 3 corr sums ---- */
  if (wave == 0) {
    const int row = l16;
    const int n   = n0 + row;
    const float inv = 1.0f / rowsumg[row];
    const float c0 = corrW[row][0] * inv;
    const float c1 = corrW[row][1] * inv;
    const float c2 = corrW[row][2] * inv;
    const float s0 = srck[((size_t)b * 3 + 0) * NDIM + n];
    const float s1 = srck[((size_t)b * 3 + 1) * NDIM + n];
    const float s2 = srck[((size_t)b * 3 + 2) * NDIM + n];
    float vals[15] = { s0*c0, s0*c1, s0*c2,
                       s1*c0, s1*c1, s1*c2,
                       s2*c0, s2*c1, s2*c2,
                       s0, s1, s2, c0, c1, c2 };
#pragma unroll
    for (int i = 0; i < 15; ++i) vals[i] = red16_sum(vals[i]);
    if (lane == 0) {
      float* pp = part + ((size_t)b * NBLOCKS + blockIdx.x) * 16;
#pragma unroll
      for (int i = 0; i < 15; ++i) pp[i] = vals[i];
      pp[15] = 0.0f;
    }
  }
}

/* ---------------- Kabsch 3x3 SVD finalize (1 thread / batch) ---------------- */
__device__ __forceinline__ void jacobi_rot(float S[3][3], float V[3][3], int p, int q)
{
  float apq = S[p][q];
  if (fabsf(apq) < 1e-18f) return;
  float tau = (S[q][q] - S[p][p]) / (2.0f * apq);
  float tt  = (tau >= 0.0f ? 1.0f : -1.0f) / (fabsf(tau) + sqrtf(1.0f + tau * tau));
  float cc  = rsqrtf(1.0f + tt * tt);
  float ss  = tt * cc;
  for (int k = 0; k < 3; ++k) {
    float skp = S[k][p], skq = S[k][q];
    S[k][p] = cc * skp - ss * skq; S[k][q] = ss * skp + cc * skq;
  }
  for (int k = 0; k < 3; ++k) {
    float spk = S[p][k], sqk = S[q][k];
    S[p][k] = cc * spk - ss * sqk; S[q][k] = ss * spk + cc * sqk;
  }
  for (int k = 0; k < 3; ++k) {
    float vkp = V[k][p], vkq = V[k][q];
    V[k][p] = cc * vkp - ss * vkq; V[k][q] = ss * vkp + cc * vkq;
  }
}

__device__ __forceinline__ float det3(const float M[3][3])
{
  return M[0][0] * (M[1][1] * M[2][2] - M[1][2] * M[2][1])
       - M[0][1] * (M[1][0] * M[2][2] - M[1][2] * M[2][0])
       + M[0][2] * (M[1][0] * M[2][1] - M[1][1] * M[2][0]);
}

__global__ void finalize_kernel(const float* __restrict__ part, float* __restrict__ out)
{
  const int b = threadIdx.x;
  if (b >= BATCH) return;

  float P[3][3] = {{0}}, S3[3] = {0}, C3[3] = {0};
  for (int blk = 0; blk < NBLOCKS; ++blk) {
    const float* pp = part + ((size_t)b * NBLOCKS + blk) * 16;
    for (int i = 0; i < 3; ++i)
      for (int j = 0; j < 3; ++j) P[i][j] += pp[i * 3 + j];
    for (int i = 0; i < 3; ++i) { S3[i] += pp[9 + i]; C3[i] += pp[12 + i]; }
  }

  const float invN = 1.0f / (float)NDIM;
  float sm[3], cm[3];
  for (int i = 0; i < 3; ++i) { sm[i] = S3[i] * invN; cm[i] = C3[i] * invN; }

  float H[3][3];
  for (int i = 0; i < 3; ++i)
    for (int j = 0; j < 3; ++j)
      H[i][j] = P[i][j] - (float)NDIM * sm[i] * cm[j];

  /* eigendecomposition of H^T H via cyclic Jacobi */
  float S[3][3];
  for (int p = 0; p < 3; ++p)
    for (int q = 0; q < 3; ++q)
      S[p][q] = H[0][p] * H[0][q] + H[1][p] * H[1][q] + H[2][p] * H[2][q];
  float V[3][3] = {{1,0,0},{0,1,0},{0,0,1}};
  for (int sw = 0; sw < 12; ++sw) {
    jacobi_rot(S, V, 0, 1);
    jacobi_rot(S, V, 0, 2);
    jacobi_rot(S, V, 1, 2);
  }
  float ev[3] = { S[0][0], S[1][1], S[2][2] };
  int i0 = 0, i1 = 1, i2 = 2, t;
  if (ev[i0] < ev[i1]) { t = i0; i0 = i1; i1 = t; }
  if (ev[i0] < ev[i2]) { t = i0; i0 = i2; i2 = t; }
  if (ev[i1] < ev[i2]) { t = i1; i1 = i2; i2 = t; }
  float Vs[3][3];
  for (int i = 0; i < 3; ++i) { Vs[i][0] = V[i][i0]; Vs[i][1] = V[i][i1]; Vs[i][2] = V[i][i2]; }

  /* U columns: u_k = normalize(H v_k); u_2 from cross product (sign folds into det fix) */
  float U0[3], U1[3], U2[3];
  for (int i = 0; i < 3; ++i) {
    U0[i] = H[i][0] * Vs[0][0] + H[i][1] * Vs[1][0] + H[i][2] * Vs[2][0];
    U1[i] = H[i][0] * Vs[0][1] + H[i][1] * Vs[1][1] + H[i][2] * Vs[2][1];
  }
  float n0 = rsqrtf(U0[0]*U0[0] + U0[1]*U0[1] + U0[2]*U0[2] + 1e-30f);
  float n1 = rsqrtf(U1[0]*U1[0] + U1[1]*U1[1] + U1[2]*U1[2] + 1e-30f);
  for (int i = 0; i < 3; ++i) { U0[i] *= n0; U1[i] *= n1; }
  U2[0] = U0[1] * U1[2] - U0[2] * U1[1];
  U2[1] = U0[2] * U1[0] - U0[0] * U1[2];
  U2[2] = U0[0] * U1[1] - U0[1] * U1[0];
  float n2 = rsqrtf(U2[0]*U2[0] + U2[1]*U2[1] + U2[2]*U2[2] + 1e-30f);
  for (int i = 0; i < 3; ++i) U2[i] *= n2;

  float M[3][3];
  for (int i = 0; i < 3; ++i)
    for (int j = 0; j < 3; ++j)
      M[i][j] = Vs[i][0] * U0[j] + Vs[i][1] * U1[j] + Vs[i][2] * U2[j];
  const float d = det3(M);

  float R[3][3];
  for (int i = 0; i < 3; ++i)
    for (int j = 0; j < 3; ++j)
      R[i][j] = Vs[i][0] * U0[j] + Vs[i][1] * U1[j] + d * Vs[i][2] * U2[j];

  float tv[3];
  for (int i = 0; i < 3; ++i)
    tv[i] = cm[i] - (R[i][0] * sm[0] + R[i][1] * sm[1] + R[i][2] * sm[2]);

  for (int i = 0; i < 3; ++i)
    for (int j = 0; j < 3; ++j)
      out[(size_t)b * 9 + i * 3 + j] = R[i][j];
  for (int i = 0; i < 3; ++i)
    out[(size_t)BATCH * 9 + (size_t)b * 3 + i] = tv[i];
}

/* ---------------- host launcher ---------------- */
extern "C" void kernel_launch(void* const* d_in, const int* in_sizes, int n_in,
                              void* d_out, int out_size, void* d_ws, size_t ws_size,
                              hipStream_t stream)
{
  const float* srcE = (const float*)d_in[0];   /* [B,512,2048] */
  const float* tgtE = (const float*)d_in[1];   /* [B,512,2048] */
  const float* srck = (const float*)d_in[2];   /* [B,3,2048]   */
  const float* tgt  = (const float*)d_in[3];   /* [B,3,2048]   */
  float* out = (float*)d_out;

  const size_t QBYTES = (size_t)BATCH * NDIM * DKDIM * sizeof(unsigned short); /* 32 MiB */
  char* ws = (char*)d_ws;
  unsigned short* Qt   = (unsigned short*)ws;
  unsigned short* Kt   = (unsigned short*)(ws + QBYTES);
  float*          prt  = (float*)(ws + 2 * QBYTES);   /* 16*128*16 f32 partials */

  dim3 tb(32, 8, 1);
  dim3 tg(NDIM / 32, DKDIM / 32, BATCH);
  transpose_cvt_kernel<<<tg, tb, 0, stream>>>(srcE, Qt);
  transpose_cvt_kernel<<<tg, tb, 0, stream>>>(tgtE, Kt);

  fused_softmax_corr_kernel<<<dim3(NBLOCKS, BATCH), 512, 0, stream>>>(
      (const bf16_t*)Qt, (const bf16_t*)Kt, tgt, srck, prt);

  finalize_kernel<<<1, 32, 0, stream>>>(prt, out);

  (void)in_sizes; (void)n_in; (void)out_size; (void)ws_size;
}
